// CNN4D_35338990911554
// MI455X (gfx1250) — compile-verified, matrix operand order checked by reading, not run on hardware
//
#include <hip/hip_runtime.h>
#include <hip/hip_bf16.h>
#include <stdint.h>

// ---------------------------------------------------------------------------
// 4D CNN (3^4 taps, SAME) as 81 shifted GEMMs on v_wmma_f32_16x16x32_bf16.
// Channels-last bf16 activations, channels padded to 32 => K=32 per tap.
// Innermost spatial axis padded 16 -> 18 (zero halo) so per-lane shifted A
// loads are unconditional. e1/e2/e3 halo handled by wave-uniform scalar
// branches that skip whole load+WMMA blocks (zero rows contribute nothing).
// Per wave: 4 output rows x 32 out channels; per t3-slice: 6 B-fragments
// (3 taps x 2 N-tiles) from LDS feed 24 WMMAs.
// Layer weights (162 KB) staged to LDS via the Tensor Data Mover.
// ---------------------------------------------------------------------------

typedef __bf16 bf16_t;
typedef __attribute__((ext_vector_type(16))) __bf16 v16bf;
typedef __attribute__((ext_vector_type(8)))  float  v8f;
typedef __attribute__((ext_vector_type(4)))  unsigned int v4u;
typedef __attribute__((ext_vector_type(8)))  int v8i_;
typedef __attribute__((ext_vector_type(4)))  int v4i_;

union Frag {
    v16bf v;
    uint4 q[2];
};

#define SPAT      65536            // 16^4
#define NB        8                // batch
#define CH        32               // padded channel count
#define PL        18               // padded innermost length (1+16+1)
#define ROWSTRIDE (PL * CH)        // 576 bf16 elems per padded row
#define ROWS      (NB * 4096)      // 32768 rows
#define ROWS_PER_WAVE 4
#define ACT_ELEMS ((size_t)ROWS * ROWSTRIDE)   // 18,874,368 bf16 per buffer
#define WPL_ELEMS 82944            // 81 taps * 2 ntiles * 32 lanes * 16 bf16
#define LDS_BYTES (WPL_ELEMS * 2)  // 165888 B per layer of packed weights

#if defined(__has_builtin)
#if __has_builtin(__builtin_amdgcn_tensor_load_to_lds) && \
    __has_builtin(__builtin_amdgcn_s_wait_tensorcnt)
#define USE_TDM 1
#endif
#endif
#ifndef USE_TDM
#define USE_TDM 0
#endif

// ---------------------------------------------------------------------------
// Pack x (N,2,16^4) f32 -> padded channels-last bf16 (halo pos & ch>=2 zero)
// ---------------------------------------------------------------------------
__global__ void pack_input_kernel(const float* __restrict__ x,
                                  bf16_t* __restrict__ act) {
    size_t i = (size_t)blockIdx.x * blockDim.x + threadIdx.x;  // over ACT_ELEMS
    if (i >= ACT_ELEMS) return;
    int ch  = (int)(i & 31);
    int t   = (int)(i >> 5);         // row*PL + p4
    int p4  = t % PL;
    int row = t / PL;
    int n   = row >> 12;
    int sp3 = row & 4095;            // (d1,d2,d3)
    float v = 0.0f;
    if (ch < 2 && p4 >= 1 && p4 <= 16)
        v = x[((size_t)n * 2 + ch) * SPAT + (size_t)sp3 * 16 + (p4 - 1)];
    act[i] = (bf16_t)v;
}

// ---------------------------------------------------------------------------
// Zero-fill a bf16 buffer (b128 granularity) - establishes actB's zero halo.
// ---------------------------------------------------------------------------
__global__ void zero_fill_kernel(uint4* __restrict__ dst, size_t nvec) {
    size_t i = (size_t)blockIdx.x * blockDim.x + threadIdx.x;
    if (i < nvec) dst[i] = uint4{0u, 0u, 0u, 0u};
}

// ---------------------------------------------------------------------------
// Pack all 6 layers' weights f32 (Cout,Cin,81) into per-lane B-fragment
// order, bf16, padded to 32x32 per tap.
// elem index: tap*1024 + ntile*512 + lane*16 + j
//   lane<16:  N=lane,     K = {0..7, 16..23}
//   lane>=16: N=lane-16,  K = {8..15, 24..31}
// ---------------------------------------------------------------------------
__global__ void pack_weights_kernel(const float* __restrict__ w0,
                                    const float* __restrict__ w1,
                                    const float* __restrict__ w2,
                                    const float* __restrict__ w3,
                                    const float* __restrict__ w4,
                                    const float* __restrict__ w5,
                                    bf16_t* __restrict__ wp) {
    int i = blockIdx.x * blockDim.x + threadIdx.x;   // over 6*WPL_ELEMS
    if (i >= 6 * WPL_ELEMS) return;
    int l    = i / WPL_ELEMS;
    int rem  = i - l * WPL_ELEMS;
    int j    = rem & 15;
    int lane = (rem >> 4) & 31;
    int ntl  = (rem >> 9) & 1;
    int tap  = rem >> 10;            // 0..80

    int k0 = (lane >> 4) * 8;
    int ci = (j < 8) ? (k0 + j) : (k0 + 16 + (j - 8));
    int co = (lane & 15) + ntl * 16;

    const float* W; int Cin, Cout;
    switch (l) {
        case 0:  W = w0; Cin = 2;  Cout = 32; break;
        case 1:  W = w1; Cin = 32; Cout = 32; break;
        case 2:  W = w2; Cin = 32; Cout = 32; break;
        case 3:  W = w3; Cin = 32; Cout = 32; break;
        case 4:  W = w4; Cin = 32; Cout = 32; break;
        default: W = w5; Cin = 32; Cout = 2;  break;
    }
    float v = 0.0f;
    if (ci < Cin && co < Cout)
        v = W[((size_t)co * Cin + ci) * 81 + tap];
    wp[i] = (bf16_t)v;
}

// ---------------------------------------------------------------------------
// One conv layer. 256 threads = 8 waves; each wave owns 4 output rows.
// Dynamic LDS holds the layer's packed weights (162 KB).
// ---------------------------------------------------------------------------
__global__ void __launch_bounds__(256, 1)
conv4d_wmma_kernel(const bf16_t* __restrict__ actIn,
                   bf16_t* __restrict__ actOut,
                   const bf16_t* __restrict__ wpLayer,
                   const float* __restrict__ slopes,
                   int layer,
                   float* __restrict__ finalOut) {
    extern __shared__ char smem[];
    const int tid = threadIdx.x;

    // ---- stage packed weights into LDS -----------------------------------
#if USE_TDM
    if (tid < 32) {                            // wave 0 drives the TDM
        const uint64_t ga  = (uint64_t)(uintptr_t)wpLayer;
        const uint32_t lds = (uint32_t)(uintptr_t)smem;   // low 32 bits = LDS offset
        const uint32_t n8  = LDS_BYTES / 8;               // 20736 x 8-byte units

        v4u g0;
        g0[0] = 1u;                                       // count=1, load descriptor
        g0[1] = lds;                                      // lds_addr
        g0[2] = (uint32_t)(ga & 0xFFFFFFFFu);             // global_addr[31:0]
        g0[3] = (uint32_t)((ga >> 32) & 0x1FFFFFFu) | (2u << 30); // addr[56:32] | type=2

        v8i_ g1;
        g1[0] = (int)(3u << 16);                          // data_size=8B; mask=0
        g1[1] = (int)((n8 & 0xFFFFu) << 16);              // tensor_dim0[15:0]
        g1[2] = (int)((n8 >> 16) | (1u << 16));           // tensor_dim0[31:16], tensor_dim1=1
        g1[3] = (int)((n8 & 0xFFFFu) << 16);              // tile_dim0
        g1[4] = 0;                                        // tile_dim1/2 unused
        g1[5] = (int)n8;                                  // tensor_dim0_stride[31:0]
        g1[6] = 0;
        g1[7] = 0;

        v4i_ gz4 = {0, 0, 0, 0};
        v8i_ gz8 = {0, 0, 0, 0, 0, 0, 0, 0};
        __builtin_amdgcn_tensor_load_to_lds(g0, g1, gz4, gz4, gz8, 0);
        __builtin_amdgcn_s_wait_tensorcnt(0);
    }
#else
    {
        const uint4* src = reinterpret_cast<const uint4*>(wpLayer);
        uint4*       dst = reinterpret_cast<uint4*>(smem);
        const int nvec = LDS_BYTES / 16;                  // 10368
        for (int i = tid; i < nvec; i += blockDim.x) dst[i] = src[i];
    }
#endif
    __syncthreads();

    const int wave = tid >> 5;
    const int lane = tid & 31;
    const int m    = lane & 15;      // A-matrix row / output position base
    const int hi   = lane >> 4;      // lane group
    const int k0   = hi * 8;         // this lane's starting K chunk

    const int grp  = blockIdx.x * 8 + wave;       // row-group id
    const int base = grp * ROWS_PER_WAVE;         // first row of group
    if (base >= ROWS) return;                     // wave-uniform
    const int n   = base >> 12;
    const int d1  = (base >> 8) & 15;
    const int d2  = (base >> 4) & 15;
    const int d3b = base & 15;       // multiple of 4; group stays in one (n,d1,d2)

    v8f c[ROWS_PER_WAVE][2];
#pragma unroll
    for (int rr = 0; rr < ROWS_PER_WAVE; ++rr) { c[rr][0] = (v8f){}; c[rr][1] = (v8f){}; }

    const uint4* lbase = reinterpret_cast<const uint4*>(smem);

    for (int t1 = 0; t1 < 3; ++t1) {
        const int e1 = d1 + t1 - 1;
        for (int t2 = 0; t2 < 3; ++t2) {
            const int e2 = d2 + t2 - 1;
            const unsigned ok12 = ((unsigned)e1 < 16u) & ((unsigned)e2 < 16u);
            for (int t3 = 0; t3 < 3; ++t3) {
                const int tapBase = ((t1 * 3 + t2) * 3 + t3) * 3;

                // ---- 6 B fragments for this t3-slice (3 taps x 2 tiles) --
                Frag b[3][2];
#pragma unroll
                for (int t4 = 0; t4 < 3; ++t4) {
#pragma unroll
                    for (int nt = 0; nt < 2; ++nt) {
                        const uint4* bp =
                            lbase + (((tapBase + t4) * 2 + nt) * 32 + lane) * 2;
                        b[t4][nt].q[0] = bp[0];
                        b[t4][nt].q[1] = bp[1];
                    }
                }

                // ---- 4 rows; skip whole block when row is in the halo ----
#pragma unroll
                for (int rr = 0; rr < ROWS_PER_WAVE; ++rr) {
                    const int e3  = d3b + rr + t3 - 1;
                    const int okr = (int)(ok12 & ((unsigned)e3 < 16u));
                    if (__builtin_amdgcn_readfirstlane(okr)) {
                        const bf16_t* rowBase = actIn +
                            (size_t)(((n * 16 + e1) * 16 + e2) * 16 + e3) * ROWSTRIDE;
#pragma unroll
                        for (int t4 = 0; t4 < 3; ++t4) {
                            Frag a;
                            const uint4* ap = reinterpret_cast<const uint4*>(
                                rowBase + (m + t4) * CH + k0);   // halo: no -1
                            a.q[0] = ap[0];                      // K = k0..k0+7
                            a.q[1] = ap[2];                      // K = k0+16..k0+23
                            c[rr][0] = __builtin_amdgcn_wmma_f32_16x16x32_bf16(
                                false, a.v, false, b[t4][0].v, (short)0,
                                c[rr][0], false, false);
                            c[rr][1] = __builtin_amdgcn_wmma_f32_16x16x32_bf16(
                                false, a.v, false, b[t4][1].v, (short)0,
                                c[rr][1], false, false);
                        }
                    }
                }
            }
        }
    }

    // ---- epilogue --------------------------------------------------------
    // C layout: VGPR r, lanes 0-15 -> (M=r, N=lane); lanes 16-31 -> (M=r+8, N=lane-16)
    if (layer < 5) {
        const float s = slopes[layer];
#pragma unroll
        for (int rr = 0; rr < ROWS_PER_WAVE; ++rr) {
            bf16_t* outRow = actOut + (size_t)(base + rr) * ROWSTRIDE;
            const int ch = lane & 15;
#pragma unroll
            for (int r = 0; r < 8; ++r) {
                float v0 = c[rr][0][r]; v0 = (v0 >= 0.0f) ? v0 : s * v0;
                float v1 = c[rr][1][r]; v1 = (v1 >= 0.0f) ? v1 : s * v1;
                const int pos = r + hi * 8 + 1;         // +1: interior of halo row
                outRow[pos * CH + ch]      = (bf16_t)v0;
                outRow[pos * CH + ch + 16] = (bf16_t)v1;
            }
        }
    } else {
        // final layer: only out channels 0,1 are real; f32 output (N,2,16^4)
        const int co = lane & 15;    // c[..][0] tile covers co 0..15
        if (co < 2) {
#pragma unroll
            for (int rr = 0; rr < ROWS_PER_WAVE; ++rr) {
                const int d3 = d3b + rr;
                const size_t obase = ((size_t)n * 2 + co) * SPAT +
                                     (size_t)(((d1 * 16 + d2) * 16 + d3) * 16);
#pragma unroll
                for (int r = 0; r < 8; ++r)
                    finalOut[obase + r + hi * 8] = c[rr][0][r];
            }
        }
    }
}

// ---------------------------------------------------------------------------
// Host-side launcher
// ---------------------------------------------------------------------------
extern "C" void kernel_launch(void* const* d_in, const int* in_sizes, int n_in,
                              void* d_out, int out_size, void* d_ws, size_t ws_size,
                              hipStream_t stream) {
    const float* x  = (const float*)d_in[0];
    const float* k0 = (const float*)d_in[1];
    const float* k1 = (const float*)d_in[2];
    const float* k2 = (const float*)d_in[3];
    const float* k3 = (const float*)d_in[4];
    const float* k4 = (const float*)d_in[5];
    const float* k5 = (const float*)d_in[6];
    const float* slopes = (const float*)d_in[7];

    // scratch layout: [0,1MB) packed weights; then two padded bf16 buffers
    bf16_t* wp   = (bf16_t*)d_ws;
    bf16_t* actA = (bf16_t*)((char*)d_ws + (1u << 20));
    bf16_t* actB = actA + ACT_ELEMS;

    // pack weights: 6 * 82944 elements
    {
        const int total = 6 * WPL_ELEMS;
        pack_weights_kernel<<<(total + 255) / 256, 256, 0, stream>>>(
            k0, k1, k2, k3, k4, k5, wp);
    }
    // pack input into actA (writes full padded buffer incl. zero halo)
    {
        const size_t total = ACT_ELEMS;
        pack_input_kernel<<<(unsigned)((total + 255) / 256), 256, 0, stream>>>(x, actA);
    }
    // zero actB once per launch to establish its halo
    {
        const size_t nvec = ACT_ELEMS * 2 / 16;
        zero_fill_kernel<<<(unsigned)((nvec + 255) / 256), 256, 0, stream>>>(
            (uint4*)actB, nvec);
    }

    // 6 conv layers, ping-pong A<->B; last layer writes f32 d_out
    bf16_t* bufs[2] = { actA, actB };
    int cur = 0;
    const int blocks = ROWS / (8 * ROWS_PER_WAVE);   // 1024
    for (int l = 0; l < 6; ++l) {
        conv4d_wmma_kernel<<<blocks, 256, LDS_BYTES, stream>>>(
            bufs[cur], bufs[cur ^ 1], wp + (size_t)l * WPL_ELEMS,
            slopes, l, (float*)d_out);
        cur ^= 1;
    }
}
